// MyGNN_3865470566850
// MI455X (gfx1250) — matrix operand order repaired and checked by reading, not compiled, not run on hardware
//
#include <hip/hip_runtime.h>
#include <cstddef>

#define N_NODES 13
#define MAX_DIM 33
#define OBS_DIM 348
#define HIDDEN  128
#define PAD_IDX 348
#define ROWS    32      // batch rows per workgroup
#define K1      64      // layer-1 K padded (33 -> 64, two wmma K-steps)

// padded LDS row strides (halves); multiples of 8 halves = 16B for aligned b128
#define SA_STRIDE  72   // sA  rows (K1 + 8)
#define SW1_STRIDE 72   // sW1 rows (K1 + 8)
#define SH_STRIDE  136  // sH  rows (HIDDEN + 8)
#define SW2_STRIDE 136  // sW2 rows (HIDDEN + 8)

typedef __attribute__((ext_vector_type(16))) _Float16 v16h;
typedef __attribute__((ext_vector_type(8)))  _Float16 v8h;
typedef __attribute__((ext_vector_type(8)))  float    v8f;

// ---------------- compile-time replica of _build_indices() ----------------
struct FeatTbl { int v[N_NODES][MAX_DIM]; };

static constexpr FeatTbl build_feat() {
  FeatTbl t{};
  for (int k = 0; k < N_NODES; ++k)
    for (int j = 0; j < MAX_DIM; ++j) t.v[k][j] = PAD_IDX;
  const int base[N_NODES][12] = {
    {0,1,2,3,4,22,23,24,25,26,27,-1},
    {5,6,28,29,-1,0,0,0,0,0,0,0},
    {7,30,-1,0,0,0,0,0,0,0,0,0},
    {8,9,10,11,31,32,33,34,-1,0,0,0},
    {11,34,-1,0,0,0,0,0,0,0,0,0},
    {-1,0,0,0,0,0,0,0,0,0,0,0},
    {12,13,14,15,35,36,37,38,-1,0,0,0},
    {15,38,-1,0,0,0,0,0,0,0,0,0},
    {-1,0,0,0,0,0,0,0,0,0,0,0},
    {16,17,18,39,40,41,-1,0,0,0,0,0},
    {18,41,-1,0,0,0,0,0,0,0,0,0},
    {19,20,21,42,43,44,-1,0,0,0,0,0},
    {21,44,-1,0,0,0,0,0,0,0,0,0},
  };
  const int qf[N_NODES][4] = {
    {-1,0,0,0},{-1,0,0,0},{0,1,2,-1},{3,4,5,-1},{6,-1,0,0},{-1,0,0,0},
    {7,8,9,-1},{10,-1,0,0},{-1,0,0,0},{11,12,-1,0},{13,-1,0,0},
    {14,15,-1,0},{16,-1,0,0},
  };
  for (int k = 0; k < N_NODES; ++k) {
    int p = 0;
    for (int i = 0; i < 12 && base[k][i] >= 0; ++i) t.v[k][p++] = base[k][i];
    for (int i = 0; i < 10; ++i) t.v[k][p++] = 45  + k * 10 + i;
    for (int i = 0; i < 6;  ++i) t.v[k][p++] = 175 + k * 6  + i;
    for (int i = 0; i < 4 && qf[k][i] >= 0; ++i) t.v[k][p++] = 253 + qf[k][i];
    for (int i = 0; i < 6;  ++i) t.v[k][p++] = 270 + k * 6  + i;
  }
  return t;
}

__constant__ FeatTbl c_feat = build_feat();

union V16 { v16h v; v8h h[2]; };

__global__ __launch_bounds__(256)
void gnn_mlp_wmma_kernel(const float* __restrict__ x,
                         const float* __restrict__ W1,
                         const float* __restrict__ b1,
                         const float* __restrict__ W2,
                         const float* __restrict__ b2,
                         float* __restrict__ out,
                         int Btot)
{
  // Phase-overlapped LDS:
  //   phase 1: sA (32x72 h = 4608B) + sW1 (128x72 h = 18432B)   = 23040B
  //   phase 2: sW2 (128x136 h = 34816B) overlays the same region
  //   persistent: sH (32x136 h = 8704B)
  __shared__ __align__(16) unsigned char smem0[34816];
  __shared__ __align__(16) unsigned char smem1[8704];
  _Float16* sA  = (_Float16*)smem0;
  _Float16* sW1 = (_Float16*)(smem0 + ROWS * SA_STRIDE * 2);
  _Float16* sW2 = (_Float16*)smem0;
  _Float16* sH  = (_Float16*)smem1;

  const int node = blockIdx.y;
  const int row0 = blockIdx.x * ROWS;
  const int tid  = threadIdx.x;
  const int nt   = tid >> 5;      // wave id == N-tile (0..7)
  const int lane = tid & 31;
  const int c    = lane >> 4;     // lane half (0/1)
  const int mr   = lane & 15;

  // ---------------- Phase 1 fill: gather x -> sA, W1^T -> sW1 -------------
  // sA[r][k] = f16( x[row0+r][feat_idx[node][k]] ), k >= 33 -> 0
  for (int e = tid; e < ROWS * K1; e += 256) {
    int k = e & 63, r = e >> 6;
    float v = 0.0f;
    int grow = row0 + r;
    if (k < MAX_DIM && grow < Btot) {
      int idx = c_feat.v[node][k];
      if (idx < OBS_DIM) v = x[(size_t)grow * OBS_DIM + idx];
    }
    sA[r * SA_STRIDE + k] = (_Float16)v;
  }
  // sW1[h][k] = f16( W1[node][k][h] )   (transpose to K-contiguous rows)
  const float* W1n = W1 + (size_t)node * MAX_DIM * HIDDEN;
  for (int e = tid; e < HIDDEN * K1; e += 256) {
    int h = e & 127, k = e >> 7;                 // coalesced over h
    float v = (k < MAX_DIM) ? W1n[k * HIDDEN + h] : 0.0f;
    sW1[h * SW1_STRIDE + k] = (_Float16)v;
  }
  __syncthreads();

  // ---------------- Layer 1: acc[mt] = xg16 * W1t16 (f32 accum) ----------
  v8f acc0 = {}, acc1 = {};
#pragma unroll
  for (int ks = 0; ks < 2; ++ks) {
    // B fragment: row = nt*16 + (lane%16), halves K = ks*32 + 8c + [0..7], then +16
    const _Float16* bp = sW1 + (nt * 16 + mr) * SW1_STRIDE + ks * 32 + 8 * c;
    V16 bf; bf.h[0] = *(const v8h*)bp; bf.h[1] = *(const v8h*)(bp + 16);
#pragma unroll
    for (int mt = 0; mt < 2; ++mt) {
      // A fragment: row = mt*16 + (lane%16), halves K = ks*32 + 8c + [0..7], then +16
      const _Float16* ap = sA + (mt * 16 + mr) * SA_STRIDE + ks * 32 + 8 * c;
      V16 af; af.h[0] = *(const v8h*)ap; af.h[1] = *(const v8h*)(ap + 16);
      if (mt == 0)
        acc0 = __builtin_amdgcn_wmma_f32_16x16x32_f16(false, af.v, false, bf.v,
                                                      (short)0, acc0, false, false);
      else
        acc1 = __builtin_amdgcn_wmma_f32_16x16x32_f16(false, af.v, false, bf.v,
                                                      (short)0, acc1, false, false);
    }
  }

  // bias + ELU, write h (f16) to sH
  {
    const float bias = b1[(size_t)node * HIDDEN + nt * 16 + mr];
#pragma unroll
    for (int r = 0; r < 8; ++r) {
      float v0 = acc0[r] + bias;
      v0 = v0 > 0.0f ? v0 : (__expf(v0) - 1.0f);
      sH[(0 * 16 + r + 8 * c) * SH_STRIDE + nt * 16 + mr] = (_Float16)v0;
      float v1 = acc1[r] + bias;
      v1 = v1 > 0.0f ? v1 : (__expf(v1) - 1.0f);
      sH[(1 * 16 + r + 8 * c) * SH_STRIDE + nt * 16 + mr] = (_Float16)v1;
    }
  }
  __syncthreads();   // sA/sW1 reads done, sH complete

  // ---------------- Phase 2 fill: W2^T -> sW2 (overlays sA/sW1) ----------
  const float* W2n = W2 + (size_t)node * HIDDEN * HIDDEN;
  for (int e = tid; e < HIDDEN * HIDDEN; e += 256) {
    int h2 = e & 127, k = e >> 7;                // coalesced over h2
    sW2[h2 * SW2_STRIDE + k] = (_Float16)W2n[k * HIDDEN + h2];
  }
  __syncthreads();

  // ---------------- Layer 2: out = h16 * W2t16 (f32 accum) ---------------
  v8f oc0 = {}, oc1 = {};
#pragma unroll
  for (int ks = 0; ks < 4; ++ks) {
    const _Float16* bp = sW2 + (nt * 16 + mr) * SW2_STRIDE + ks * 32 + 8 * c;
    V16 bf; bf.h[0] = *(const v8h*)bp; bf.h[1] = *(const v8h*)(bp + 16);
#pragma unroll
    for (int mt = 0; mt < 2; ++mt) {
      const _Float16* ap = sH + (mt * 16 + mr) * SH_STRIDE + ks * 32 + 8 * c;
      V16 af; af.h[0] = *(const v8h*)ap; af.h[1] = *(const v8h*)(ap + 16);
      if (mt == 0)
        oc0 = __builtin_amdgcn_wmma_f32_16x16x32_f16(false, af.v, false, bf.v,
                                                     (short)0, oc0, false, false);
      else
        oc1 = __builtin_amdgcn_wmma_f32_16x16x32_f16(false, af.v, false, bf.v,
                                                     (short)0, oc1, false, false);
    }
  }

  // bias + store: out[(grow*13 + node)*128 + col]
  {
    const float bias = b2[(size_t)node * HIDDEN + nt * 16 + mr];
    const int col = nt * 16 + mr;
#pragma unroll
    for (int r = 0; r < 8; ++r) {
      int g0 = row0 + 0 * 16 + r + 8 * c;
      if (g0 < Btot)
        out[((size_t)g0 * N_NODES + node) * HIDDEN + col] = oc0[r] + bias;
      int g1 = row0 + 1 * 16 + r + 8 * c;
      if (g1 < Btot)
        out[((size_t)g1 * N_NODES + node) * HIDDEN + col] = oc1[r] + bias;
    }
  }
}

extern "C" void kernel_launch(void* const* d_in, const int* in_sizes, int n_in,
                              void* d_out, int out_size, void* d_ws, size_t ws_size,
                              hipStream_t stream) {
  const float* x  = (const float*)d_in[0];
  const float* W1 = (const float*)d_in[1];
  const float* b1 = (const float*)d_in[2];
  const float* W2 = (const float*)d_in[3];
  const float* b2 = (const float*)d_in[4];
  float* out = (float*)d_out;

  const int B = in_sizes[0] / OBS_DIM;
  dim3 grid((B + ROWS - 1) / ROWS, N_NODES);
  gnn_mlp_wmma_kernel<<<grid, dim3(256), 0, stream>>>(x, W1, b1, W2, b2, out, B);
}
